// OTKernel_68315749810576
// MI455X (gfx1250) — matrix operand order repaired and checked by reading, not compiled, not run on hardware
//
#include <hip/hip_runtime.h>
#include <hip/hip_bf16.h>

// ---------------------------------------------------------------------------
// Optimal-transport attention (Sinkhorn) for MI455X / gfx1250.
//
// Strategy: one workgroup per (batch, head). Each sub-problem's K matrix
// (2048 x 64) is kept resident in LDS as bf16 (256 KB < 320 KB WGP LDS)
// across: WMMA gemm -> exp -> 10 Sinkhorn iterations -> rescale -> WMMA gemm.
// HBM traffic is limited to x (bf16, read twice, L2-shared by 4 heads) and
// the 8.4 MB f32 output.
// ---------------------------------------------------------------------------

typedef __attribute__((ext_vector_type(16))) __bf16 v16bf;
typedef __attribute__((ext_vector_type(8)))  __bf16 v8bf;
typedef __attribute__((ext_vector_type(8)))  float  v8f;

#define BATCH    64
#define I_SIZE   2048
#define D_DIM    128
#define HEADS    4
#define O_SIZE   64
#define EPS_INV  20.0f          // 1/0.05
#define NORM_EPS 1e-6f

// LDS layout (bytes). KT stride chosen: multiple of 8 halves (16B-aligned
// v8bf chunks) and stride/2 = 1028 dwords == 4 (mod 64 banks) -> only 2-4 way
// conflicts on strided row accesses.
#define KT_STRIDE   2056                       // halves per Kt row (>= 2048)
#define WNT_STRIDE  72                         // half the WnT row stride; rows are 2*72 = 144 halves (>= 128)
#define LDS_KT      0
#define LDS_WNT     (LDS_KT  + O_SIZE * KT_STRIDE * 2)        // 263168
#define LDS_U       (LDS_WNT + D_DIM * WNT_STRIDE * 2)        // +18432
#define LDS_RNI     (LDS_U   + I_SIZE * 4)                    // +8192
#define LDS_V       (LDS_RNI + I_SIZE * 4)                    // +8192
#define LDS_VPART   (LDS_V   + O_SIZE * 4)                    // +256
#define LDS_TOTAL   (LDS_VPART + 4 * O_SIZE * 4)              // = 299264 B

// ---------------------------------------------------------------------------
// Prep 1: x f32 -> bf16, plus per-row inverse clamped L2 norm. 1 wave / row.
// ---------------------------------------------------------------------------
__global__ __launch_bounds__(256) void prep_x_kernel(
    const float* __restrict__ x, __bf16* __restrict__ xbf,
    float* __restrict__ rni, int nrows) {
  int wave = threadIdx.x >> 5, lane = threadIdx.x & 31;
  int row = blockIdx.x * 8 + wave;
  if (row >= nrows) return;
  const float4 v = *(const float4*)(x + (size_t)row * D_DIM + lane * 4);
  float s = v.x * v.x + v.y * v.y + v.z * v.z + v.w * v.w;
  #pragma unroll
  for (int off = 16; off; off >>= 1) s += __shfl_xor(s, off, 32);
  if (lane == 0) rni[row] = 1.0f / fmaxf(sqrtf(s), NORM_EPS);
  union { __bf16 h[4]; uint2 u; } p;
  p.h[0] = (__bf16)v.x; p.h[1] = (__bf16)v.y;
  p.h[2] = (__bf16)v.z; p.h[3] = (__bf16)v.w;
  *(uint2*)(xbf + (size_t)row * D_DIM + lane * 4) = p.u;
}

// ---------------------------------------------------------------------------
// Prep 2: weight rows L2-normalized -> bf16. 1 wave / row (256 rows total).
// ---------------------------------------------------------------------------
__global__ __launch_bounds__(256) void prep_w_kernel(
    const float* __restrict__ w, __bf16* __restrict__ wn, int nrows) {
  int wave = threadIdx.x >> 5, lane = threadIdx.x & 31;
  int row = blockIdx.x * 8 + wave;
  if (row >= nrows) return;
  const float4 v = *(const float4*)(w + (size_t)row * D_DIM + lane * 4);
  float s = v.x * v.x + v.y * v.y + v.z * v.z + v.w * v.w;
  #pragma unroll
  for (int off = 16; off; off >>= 1) s += __shfl_xor(s, off, 32);
  float inv = 1.0f / fmaxf(sqrtf(s), NORM_EPS);
  union { __bf16 h[4]; uint2 u; } p;
  p.h[0] = (__bf16)(v.x * inv); p.h[1] = (__bf16)(v.y * inv);
  p.h[2] = (__bf16)(v.z * inv); p.h[3] = (__bf16)(v.w * inv);
  *(uint2*)(wn + (size_t)row * D_DIM + lane * 4) = p.u;
}

// ---------------------------------------------------------------------------
// Main kernel: one workgroup (8 wave32s) per (batch, head).
// ---------------------------------------------------------------------------
__global__ __launch_bounds__(256) void ot_main_kernel(
    const __bf16* __restrict__ xbf, const float* __restrict__ rniG,
    const __bf16* __restrict__ wn, float* __restrict__ out) {
  extern __shared__ char smem[];
  __bf16* Kt    = (__bf16*)(smem + LDS_KT);     // [O_SIZE][KT_STRIDE] bf16
  __bf16* WnT   = (__bf16*)(smem + LDS_WNT);    // [O_SIZE][2*WNT_STRIDE] bf16
  float*  uL    = (float*)(smem + LDS_U);       // [I_SIZE]
  float*  rni   = (float*)(smem + LDS_RNI);     // [I_SIZE]
  float*  vL    = (float*)(smem + LDS_V);       // [O_SIZE]
  float*  vpart = (float*)(smem + LDS_VPART);   // [4][O_SIZE]

  const int bh = blockIdx.x;
  const int b = bh >> 2, h = bh & 3;
  const __bf16* xb = xbf + (size_t)b * I_SIZE * D_DIM;
  const float*  rg = rniG + (size_t)b * I_SIZE;
  const __bf16* wh = wn + (size_t)h * O_SIZE * D_DIM;
  float* outp = out + (size_t)b * O_SIZE * HEADS * D_DIM + (size_t)h * D_DIM;

  const int tid = threadIdx.x;
  const int lane = tid & 31, wave = tid >> 5;
  const int lmod = lane & 15;
  const int hi16 = (lane < 16) ? 0 : 8;

  // ---- Phase 0: stage wn rows (o-major, d contiguous, padded) into LDS; cache rni; init v ----
  for (int e = tid; e < O_SIZE * D_DIM; e += 256) {
    int o = e >> 7, d = e & 127;
    WnT[o * (2 * WNT_STRIDE) + d] = wh[e];
  }
  for (int i = tid; i < I_SIZE; i += 256) rni[i] = rg[i];
  if (tid < O_SIZE) vL[tid] = 1.0f;
  __syncthreads();

  // ---- Phase 1: Kt[o][i] = exp( (x_i . wn_o) * rni[i] / eps ), via WMMA ----
  // A = x tile (M=i 16 x K=d 32), B = wnT tile (K=d 32 x N=o 16), D = i x o.
  {
    const int nt = wave & 3;            // o-tile index (4 tiles)
    const int n0 = nt * 16;
    for (int mt = (wave >> 2); mt < I_SIZE / 16; mt += 2) {   // 128 i-tiles
      const int i0 = mt * 16;
      const __bf16* xrow = xb + (size_t)(i0 + lmod) * D_DIM;
      v8f c = {};
      #pragma unroll
      for (int ks = 0; ks < 4; ++ks) {
        const int k0 = ks * 32;
        // A lane layout: halves [base, base+8) and [base+16, base+24)
        v8bf a0 = *(const v8bf*)(xrow + k0 + hi16);
        v8bf a1 = *(const v8bf*)(xrow + k0 + hi16 + 16);
        v16bf A = __builtin_shufflevector(a0, a1, 0, 1, 2, 3, 4, 5, 6, 7,
                                          8, 9, 10, 11, 12, 13, 14, 15);
        // B lane layout: column o = n0 + lane%16, halves [base, base+8) and [base+16, base+24)
        const __bf16* bp = WnT + (n0 + lmod) * (2 * WNT_STRIDE) + k0 + hi16;
        v8bf b0 = *(const v8bf*)(bp);
        v8bf b1 = *(const v8bf*)(bp + 16);
        v16bf Bv = __builtin_shufflevector(b0, b1, 0, 1, 2, 3, 4, 5, 6, 7,
                                           8, 9, 10, 11, 12, 13, 14, 15);
        c = __builtin_amdgcn_wmma_f32_16x16x32_bf16(
            false, A, false, Bv, (short)0, c, false, false);
      }
      // D layout: N (=o) = lane%16, M (=i) = vgpr + (lane<16 ? 0 : 8).
      const int o  = n0 + lmod;
      const int ib = i0 + hi16;
      const float* rp = rni + ib;       // 8 contiguous rni values per lane
      v8bf kv;
      #pragma unroll
      for (int r = 0; r < 8; ++r)
        kv[r] = (__bf16)__expf(c[r] * rp[r] * EPS_INV);
      *(v8bf*)(Kt + (size_t)o * KT_STRIDE + ib) = kv;   // 1x ds_store_b128
    }
  }
  __syncthreads();

  // ---- Phase 2: Sinkhorn, 10 iterations fully in LDS ----
  const float a_const = (float)O_SIZE / (float)I_SIZE;
  for (int it = 0; it < 10; ++it) {
    {   // u_i = a / sum_o Kt[o][i] * v[o]  (8 contiguous i per thread)
      const int i0 = tid * 8;
      v8f s = {};
      for (int o = 0; o < O_SIZE; ++o) {
        v8bf kv = *(const v8bf*)(Kt + (size_t)o * KT_STRIDE + i0);
        float vo = vL[o];
        #pragma unroll
        for (int e = 0; e < 8; ++e) s[e] += (float)kv[e] * vo;
      }
      v8f uo;
      #pragma unroll
      for (int e = 0; e < 8; ++e) uo[e] = a_const / s[e];
      *(v8f*)(uL + i0) = uo;
    }
    __syncthreads();
    {   // v_o = 1 / sum_i u_i * Kt[o][i]  (4 segments x 64 columns)
      const int o = tid & 63, seg = tid >> 6;
      float s = 0.f;
      const __bf16* krow = Kt + (size_t)o * KT_STRIDE;
      for (int i = seg * 512; i < seg * 512 + 512; i += 8) {
        v8bf kv = *(const v8bf*)(krow + i);
        v8f  uv = *(const v8f*)(uL + i);
        #pragma unroll
        for (int e = 0; e < 8; ++e) s += (float)kv[e] * uv[e];
      }
      vpart[seg * 64 + o] = s;
    }
    __syncthreads();
    if (tid < O_SIZE) {
      float s = vpart[tid] + vpart[64 + tid] + vpart[128 + tid] + vpart[192 + tid];
      vL[tid] = 1.0f / s;
    }
    __syncthreads();
  }

  // ---- Phase 2.5: Kt[o][i] *= u_i * v_o (attn, kept transposed for WMMA-A) --
  for (int idx = tid; idx < O_SIZE * (I_SIZE / 8); idx += 256) {
    const int o = idx >> 8;
    const int i0 = (idx & 255) * 8;
    __bf16* kp = Kt + (size_t)o * KT_STRIDE + i0;
    v8bf kv = *(v8bf*)kp;
    v8f  uv = *(const v8f*)(uL + i0);
    const float vo = vL[o];
    #pragma unroll
    for (int e = 0; e < 8; ++e) kv[e] = (__bf16)((float)kv[e] * uv[e] * vo);
    *(v8bf*)kp = kv;
  }
  __syncthreads();

  // ---- Phase 3: out[o][d] = sum_i attn[o][i] * x[i][d], via WMMA ----
  // A = Kt tile (M=o 16 x K=i 32) straight from LDS; B = x rows (K=i 32 x N=d 16).
  {
    const int mo = wave & 3;            // o-tile (4)
    const int npart = wave >> 2;        // d half (0: d<64, 1: d>=64)
    const int orow = mo * 16 + lmod;
    const __bf16* arow = Kt + (size_t)orow * KT_STRIDE;
    v8f acc0 = {}, acc1 = {}, acc2 = {}, acc3 = {};
    for (int k0 = 0; k0 < I_SIZE; k0 += 32) {
      v8bf a0 = *(const v8bf*)(arow + k0 + hi16);
      v8bf a1 = *(const v8bf*)(arow + k0 + hi16 + 16);
      v16bf A = __builtin_shufflevector(a0, a1, 0, 1, 2, 3, 4, 5, 6, 7,
                                        8, 9, 10, 11, 12, 13, 14, 15);
      const __bf16* xk = xb + (size_t)(k0 + lane) * D_DIM + npart * 64;
      __builtin_prefetch(xk + 32 * D_DIM, 0, 1);      // global_prefetch next tile
      v16bf B0, B1, B2, B3;
      #pragma unroll
      for (int s = 0; s < 16; ++s) {
        const int kk = k0 + hi16 + (s < 8 ? s : s + 8);
        const __bf16* xp = xb + (size_t)kk * D_DIM + npart * 64 + lmod;
        B0[s] = xp[0]; B1[s] = xp[16]; B2[s] = xp[32]; B3[s] = xp[48];
      }
      acc0 = __builtin_amdgcn_wmma_f32_16x16x32_bf16(false, A, false, B0, (short)0, acc0, false, false);
      acc1 = __builtin_amdgcn_wmma_f32_16x16x32_bf16(false, A, false, B1, (short)0, acc1, false, false);
      acc2 = __builtin_amdgcn_wmma_f32_16x16x32_bf16(false, A, false, B2, (short)0, acc2, false, false);
      acc3 = __builtin_amdgcn_wmma_f32_16x16x32_bf16(false, A, false, B3, (short)0, acc3, false, false);
    }
    // D layout: N (=d) = lane%16, M (=o) = vgpr + (lane<16 ? 0 : 8).
    const int d0 = npart * 64 + lmod;
    const int ob = mo * 16 + hi16;
    #pragma unroll
    for (int r = 0; r < 8; ++r) {
      float* op = outp + (size_t)(ob + r) * (HEADS * D_DIM) + d0;
      op[0]  = acc0[r];
      op[16] = acc1[r];
      op[32] = acc2[r];
      op[48] = acc3[r];
    }
  }
}

// ---------------------------------------------------------------------------
// Host launcher. Workspace: xbf (33.5 MB) + rni (0.5 MB) + wn (64 KB).
// ---------------------------------------------------------------------------
extern "C" void kernel_launch(void* const* d_in, const int* in_sizes, int n_in,
                              void* d_out, int out_size, void* d_ws, size_t ws_size,
                              hipStream_t stream) {
  const float* x = (const float*)d_in[0];   // [64, 2048, 128] f32
  const float* w = (const float*)d_in[1];   // [4, 64, 128] f32
  float* out = (float*)d_out;               // [64, 64, 512] f32

  char* ws = (char*)d_ws;
  const size_t xbf_bytes = (size_t)BATCH * I_SIZE * D_DIM * 2;   // 33,554,432
  const size_t rni_bytes = (size_t)BATCH * I_SIZE * 4;           //    524,288
  __bf16* xbf = (__bf16*)ws;
  float*  rni = (float*)(ws + xbf_bytes);
  __bf16* wnb = (__bf16*)(ws + xbf_bytes + rni_bytes);           // +65,536

  const int xrows = BATCH * I_SIZE;          // 131072
  prep_x_kernel<<<xrows / 8, 256, 0, stream>>>(x, xbf, rni, xrows);
  prep_w_kernel<<<(HEADS * O_SIZE) / 8, 256, 0, stream>>>(w, wnb, HEADS * O_SIZE);
  ot_main_kernel<<<BATCH * HEADS, 256, LDS_TOTAL, stream>>>(xbf, rni, wnb, out);
}
